// LTDistributedAttention_11089605558983
// MI455X (gfx1250) — matrix operand order repaired and checked by reading, not compiled, not run on hardware
//
#include <hip/hip_runtime.h>

typedef __attribute__((ext_vector_type(16))) _Float16 v16h;
typedef __attribute__((ext_vector_type(8)))  _Float16 v8h;
typedef __attribute__((ext_vector_type(8)))  float    v8f;
typedef __attribute__((ext_vector_type(4)))  int      i4;

#define SEQ    2048
#define NH     32
#define DH     128
#define QCHUNK 128
#define KTILE  32

// ---------------- CDNA5 async-to-LDS plumbing (guarded) ----------------
#if defined(__HIP_DEVICE_COMPILE__) && __has_builtin(__builtin_amdgcn_global_load_async_to_lds_b128)
#define USE_ASYNC_LDS 1
#else
#define USE_ASYNC_LDS 0
#endif

__device__ __forceinline__ void copy_b128_to_lds(const _Float16* g, _Float16* l) {
#if USE_ASYNC_LDS
  __builtin_amdgcn_global_load_async_to_lds_b128(
      (__attribute__((address_space(1))) i4*)g,
      (__attribute__((address_space(3))) i4*)l, 0, 0);
#else
  *(uint4*)l = *(const uint4*)g;
#endif
}

__device__ __forceinline__ void wait_async_copies() {
#if USE_ASYNC_LDS
#if __has_builtin(__builtin_amdgcn_s_wait_asynccnt)
  __builtin_amdgcn_s_wait_asynccnt(0);
#else
  asm volatile("s_wait_asynccnt 0" ::: "memory");
#endif
#endif
}

// ---------------- pre-pass: kv(f32) -> K f16 [h][s][d], V^T f16 [h][d][s] ----------------
__global__ __launch_bounds__(256)
void kv_prepass(const float* __restrict__ kv, _Float16* __restrict__ kh,
                _Float16* __restrict__ vT) {
  __shared__ float lds_v[32][129];
  const int head = blockIdx.y;
  const int sblk = blockIdx.x * 32;
  const int tid  = threadIdx.x;

  // K: straight convert, 8-element (16B) chunks, coalesced both sides
#pragma unroll
  for (int j = 0; j < 2; ++j) {
    int ch = tid + j * 256;            // 0..511 over (key 0..31, dgroup 0..15)
    int k = ch >> 4, g = ch & 15;
    const float* src = kv + (((size_t)(sblk + k) * 2 + 0) * NH + head) * DH + g * 8;
    v8h t;
#pragma unroll
    for (int e = 0; e < 8; ++e) t[e] = (_Float16)src[e];
    *(v8h*)(kh + ((size_t)head * SEQ + sblk + k) * DH + g * 8) = t;
  }
  // V: tile-transpose through LDS
#pragma unroll
  for (int i = 0; i < 16; ++i) {
    int idx = tid + i * 256;
    int k = idx >> 7, d = idx & 127;
    lds_v[k][d] = kv[(((size_t)(sblk + k) * 2 + 1) * NH + head) * DH + d];
  }
  __syncthreads();
  {
    int d = tid >> 1, sh = (tid & 1) * 16;
    v8h t0, t1;
#pragma unroll
    for (int e = 0; e < 8; ++e) t0[e] = (_Float16)lds_v[sh + e][d];
#pragma unroll
    for (int e = 0; e < 8; ++e) t1[e] = (_Float16)lds_v[sh + 8 + e][d];
    _Float16* dst = vT + ((size_t)head * DH + d) * SEQ + sblk + sh;
    *(v8h*)(dst)     = t0;
    *(v8h*)(dst + 8) = t1;
  }
}

// ---------------- main: flash attention, async double-buffered staging ----------------
// S^T = K*Q^T (C layout: q-row per lane -> cheap softmax), O^T = V^T*P^T.
__global__ __launch_bounds__(256)
void fa_causal_wmma_async(const float* __restrict__ q,
                          const _Float16* __restrict__ kh,
                          const _Float16* __restrict__ vT,
                          float* __restrict__ out) {
  // double-buffered A-fragment-ordered tiles
  __shared__ _Float16 lk[2][4 * 2 * 32 * 16];   // 2 x 8 KB
  __shared__ _Float16 lv[2][8 * 32 * 16];       // 2 x 8 KB

  const int head      = blockIdx.y;
  const int wg_qbase  = blockIdx.x * QCHUNK;
  const int tid       = threadIdx.x;
  const int lane      = tid & 31;
  const int wv        = tid >> 5;
  const int laneHi    = lane >> 4;
  const int qs        = wg_qbase + wv * 16 + (lane & 15);
  const int wave_qmax = wg_qbase + wv * 16 + 15;

  const float scale = 0.08838834764831845f;     // 1/sqrt(128)

  const _Float16* khh = kh + (size_t)head * SEQ * DH;
  const _Float16* vth = vT + (size_t)head * DH * SEQ;

  // Q fragments (B-layout of S^T gemm), 4 d-chunks of 32
  v16h qb[4];
#pragma unroll
  for (int c = 0; c < 4; ++c) {
    const float* qp = q + ((size_t)qs * NH + head) * DH + c * 32 + laneHi * 16;
#pragma unroll
    for (int h = 0; h < 16; ++h) qb[c][h] = (_Float16)(qp[h] * scale);
  }

  v8f o[8];
#pragma unroll
  for (int cc = 0; cc < 8; ++cc) o[cc] = (v8f){};
  float mrow = -INFINITY;
  float lrow = 0.0f;

  const int nkt = (wg_qbase + QCHUNK) / KTILE;

  // stage tile kt into buffer b: 2 K-chunks + 2 V-chunks (16B each) per thread
  auto stage = [&](int kt, int b) {
    const _Float16* kp = khh + (size_t)(kt * KTILE) * DH;
    const _Float16* vp = vth + kt * KTILE;
#pragma unroll
    for (int j = 0; j < 2; ++j) {
      int ch = tid + j * 256;                      // 0..511
      {                                            // K: (key k, dgroup g of 8)
        int k = ch >> 4, g = ch & 15;
        int off = (((g >> 2) * 2 + (k >> 4)) * 32 + (k & 15) + 16 * ((g >> 1) & 1)) * 16
                  + 8 * (g & 1);
        copy_b128_to_lds(kp + (size_t)k * DH + g * 8, &lk[b][off]);
      }
      {                                            // V^T: (d, keygroup kg of 8)
        int d = ch >> 2, kg = ch & 3;
        int off = ((d >> 4) * 32 + (d & 15) + 16 * (kg >> 1)) * 16 + 8 * (kg & 1);
        copy_b128_to_lds(vp + (size_t)d * SEQ + kg * 8, &lv[b][off]);
      }
    }
  };

  stage(0, 0);
  int buf = 0;

  for (int kt = 0; kt < nkt; ++kt) {
    const int kb = kt * KTILE;
    wait_async_copies();
    __syncthreads();                              // tile kt landed in lk/lv[buf]
    if (kt + 1 < nkt) stage(kt + 1, buf ^ 1);     // overlap next DMA with WMMA

    if (kb <= wave_qmax) {                        // wave-uniform: EXEC all ones
      const _Float16* lkb = &lk[buf][0];
      const _Float16* lvb = &lv[buf][0];
      // ---- S^T = K * Q^T : two 16(key)x16(q) tiles ----
      v8f c0 = (v8f){}, c1 = (v8f){};
#pragma unroll
      for (int c = 0; c < 4; ++c) {
        v16h a0 = *(const v16h*)&lkb[((c * 2 + 0) * 32 + lane) << 4];
        v16h a1 = *(const v16h*)&lkb[((c * 2 + 1) * 32 + lane) << 4];
        c0 = __builtin_amdgcn_wmma_f32_16x16x32_f16(false, a0, false, qb[c], (short)0, c0, false, false);
        c1 = __builtin_amdgcn_wmma_f32_16x16x32_f16(false, a1, false, qb[c], (short)0, c1, false, false);
      }
      // causal mask (C: VGPR r holds key = kb + t*16 + r + 8*laneHi)
#pragma unroll
      for (int r = 0; r < 8; ++r) {
        int key0 = kb + r + 8 * laneHi;
        if (key0 > qs)      c0[r] = -INFINITY;
        if (key0 + 16 > qs) c1[r] = -INFINITY;
      }
      // online softmax per q-row
      float smax = -INFINITY;
#pragma unroll
      for (int r = 0; r < 8; ++r) smax = fmaxf(smax, fmaxf(c0[r], c1[r]));
      smax = fmaxf(smax, __shfl_xor(smax, 16, 32));
      float mnew = fmaxf(mrow, smax);
      float corr = __expf(mrow - mnew);
      float rsum = 0.0f;
#pragma unroll
      for (int r = 0; r < 8; ++r) {
        c0[r] = __expf(c0[r] - mnew);
        c1[r] = __expf(c1[r] - mnew);
        rsum += c0[r] + c1[r];
      }
      rsum += __shfl_xor(rsum, 16, 32);
      lrow = lrow * corr + rsum;
      mrow = mnew;
#pragma unroll
      for (int cc = 0; cc < 8; ++cc)
#pragma unroll
        for (int r = 0; r < 8; ++r) o[cc][r] *= corr;

      // P^T: C layout -> B layout (one shfl_xor(16) per C register)
      v16h pb = (v16h){};
#pragma unroll
      for (int r = 0; r < 8; ++r) {
        float oth0 = __shfl_xor(c0[r], 16, 32);
        float oth1 = __shfl_xor(c1[r], 16, 32);
        float lo = laneHi ? oth1  : c0[r];
        float hi = laneHi ? c1[r] : oth0;
        pb[r]     = (_Float16)lo;
        pb[r + 8] = (_Float16)hi;
      }
      // O^T += V^T * P^T
#pragma unroll
      for (int cc = 0; cc < 8; ++cc) {
        v16h va = *(const v16h*)&lvb[(cc * 32 + lane) << 4];
        o[cc] = __builtin_amdgcn_wmma_f32_16x16x32_f16(false, va, false, pb, (short)0, o[cc], false, false);
      }
    }
    __syncthreads();
    buf ^= 1;
  }

  float inv = 1.0f / lrow;
#pragma unroll
  for (int cc = 0; cc < 8; ++cc) {
    float* op = out + ((size_t)qs * NH + head) * DH + cc * 16 + laneHi * 8;
    float4 u0 = make_float4(o[cc][0] * inv, o[cc][1] * inv, o[cc][2] * inv, o[cc][3] * inv);
    float4 u1 = make_float4(o[cc][4] * inv, o[cc][5] * inv, o[cc][6] * inv, o[cc][7] * inv);
    *(float4*)(op)     = u0;
    *(float4*)(op + 4) = u1;
  }
}

// ---------------- fallback (no workspace): round-1 self-contained kernel ----------------
__global__ __launch_bounds__(256)
void fa_causal_wmma_fallback(const float* __restrict__ q,
                             const float* __restrict__ kv,
                             float* __restrict__ out) {
  __shared__ _Float16 lk[4 * 2 * 32 * 16];
  __shared__ _Float16 lv[8 * 32 * 16];
  const int head      = blockIdx.y;
  const int wg_qbase  = blockIdx.x * QCHUNK;
  const int tid       = threadIdx.x;
  const int lane      = tid & 31;
  const int wv        = tid >> 5;
  const int laneHi    = lane >> 4;
  const int qs        = wg_qbase + wv * 16 + (lane & 15);
  const int wave_qmax = wg_qbase + wv * 16 + 15;
  const float scale = 0.08838834764831845f;

  v16h qb[4];
#pragma unroll
  for (int c = 0; c < 4; ++c) {
    const float* qp = q + ((size_t)qs * NH + head) * DH + c * 32 + laneHi * 16;
#pragma unroll
    for (int h = 0; h < 16; ++h) qb[c][h] = (_Float16)(qp[h] * scale);
  }
  v8f o[8];
#pragma unroll
  for (int cc = 0; cc < 8; ++cc) o[cc] = (v8f){};
  float mrow = -INFINITY, lrow = 0.0f;
  const int nkt = (wg_qbase + QCHUNK) / KTILE;

  for (int kt = 0; kt < nkt; ++kt) {
    const int kb = kt * KTILE;
#pragma unroll
    for (int i = 0; i < 16; ++i) {
      int idx = tid + i * 256;
      int k = idx >> 7, d = idx & 127;
      float kval = kv[(((size_t)(kb + k) * 2 + 0) * NH + head) * DH + d];
      float vval = kv[(((size_t)(kb + k) * 2 + 1) * NH + head) * DH + d];
      { int c = d >> 5, doff = d & 31, t = k >> 4;
        lk[(((c * 2 + t) * 32 + (k & 15) + 16 * (doff >> 4)) << 4)
           + (doff & 7) + 8 * ((doff >> 3) & 1)] = (_Float16)kval; }
      { int cc = d >> 4;
        lv[(((cc * 32) + (d & 15) + 16 * (k >> 4)) << 4)
           + (k & 7) + 8 * ((k >> 3) & 1)] = (_Float16)vval; }
    }
    __syncthreads();
    if (kb <= wave_qmax) {
      v8f c0 = (v8f){}, c1 = (v8f){};
#pragma unroll
      for (int c = 0; c < 4; ++c) {
        v16h a0 = *(const v16h*)&lk[((c * 2 + 0) * 32 + lane) << 4];
        v16h a1 = *(const v16h*)&lk[((c * 2 + 1) * 32 + lane) << 4];
        c0 = __builtin_amdgcn_wmma_f32_16x16x32_f16(false, a0, false, qb[c], (short)0, c0, false, false);
        c1 = __builtin_amdgcn_wmma_f32_16x16x32_f16(false, a1, false, qb[c], (short)0, c1, false, false);
      }
#pragma unroll
      for (int r = 0; r < 8; ++r) {
        int key0 = kb + r + 8 * laneHi;
        if (key0 > qs)      c0[r] = -INFINITY;
        if (key0 + 16 > qs) c1[r] = -INFINITY;
      }
      float smax = -INFINITY;
#pragma unroll
      for (int r = 0; r < 8; ++r) smax = fmaxf(smax, fmaxf(c0[r], c1[r]));
      smax = fmaxf(smax, __shfl_xor(smax, 16, 32));
      float mnew = fmaxf(mrow, smax);
      float corr = __expf(mrow - mnew);
      float rsum = 0.0f;
#pragma unroll
      for (int r = 0; r < 8; ++r) {
        c0[r] = __expf(c0[r] - mnew);
        c1[r] = __expf(c1[r] - mnew);
        rsum += c0[r] + c1[r];
      }
      rsum += __shfl_xor(rsum, 16, 32);
      lrow = lrow * corr + rsum;
      mrow = mnew;
#pragma unroll
      for (int cc = 0; cc < 8; ++cc)
#pragma unroll
        for (int r = 0; r < 8; ++r) o[cc][r] *= corr;
      v16h pb = (v16h){};
#pragma unroll
      for (int r = 0; r < 8; ++r) {
        float oth0 = __shfl_xor(c0[r], 16, 32);
        float oth1 = __shfl_xor(c1[r], 16, 32);
        float lo = laneHi ? oth1  : c0[r];
        float hi = laneHi ? c1[r] : oth0;
        pb[r] = (_Float16)lo; pb[r + 8] = (_Float16)hi;
      }
#pragma unroll
      for (int cc = 0; cc < 8; ++cc) {
        v16h va = *(const v16h*)&lv[(cc * 32 + lane) << 4];
        o[cc] = __builtin_amdgcn_wmma_f32_16x16x32_f16(false, va, false, pb, (short)0, o[cc], false, false);
      }
    }
    __syncthreads();
  }
  float inv = 1.0f / lrow;
#pragma unroll
  for (int cc = 0; cc < 8; ++cc) {
    float* op = out + ((size_t)qs * NH + head) * DH + cc * 16 + laneHi * 8;
    float4 u0 = make_float4(o[cc][0] * inv, o[cc][1] * inv, o[cc][2] * inv, o[cc][3] * inv);
    float4 u1 = make_float4(o[cc][4] * inv, o[cc][5] * inv, o[cc][6] * inv, o[cc][7] * inv);
    *(float4*)(op) = u0; *(float4*)(op + 4) = u1;
  }
}

extern "C" void kernel_launch(void* const* d_in, const int* in_sizes, int n_in,
                              void* d_out, int out_size, void* d_ws, size_t ws_size,
                              hipStream_t stream) {
  const float* q  = (const float*)d_in[0];
  const float* kv = (const float*)d_in[1];
  float* out      = (float*)d_out;

  const size_t halfKV = (size_t)NH * SEQ * DH * sizeof(_Float16);   // 16.78 MB
  if (d_ws != nullptr && ws_size >= 2 * halfKV) {
    _Float16* kh = (_Float16*)d_ws;
    _Float16* vT = (_Float16*)((char*)d_ws + halfKV);
    kv_prepass<<<dim3(SEQ / 32, NH), 256, 0, stream>>>(kv, kh, vT);
    fa_causal_wmma_async<<<dim3(SEQ / QCHUNK, NH), 256, 0, stream>>>(q, kh, vT, out);
  } else {
    fa_causal_wmma_fallback<<<dim3(SEQ / QCHUNK, NH), 256, 0, stream>>>(q, kv, out);
  }
}
